// SelfAttention1d_81655918231900
// MI455X (gfx1250) — compile-verified, matrix operand order checked by reading, not run on hardware
//
#include <hip/hip_runtime.h>
#include <hip/hip_bf16.h>

typedef __attribute__((ext_vector_type(16))) _Float16 v16h;
typedef __attribute__((ext_vector_type(8)))  _Float16 v8h;
typedef __attribute__((ext_vector_type(8)))  float    v8f;

#define C_DIM 512
#define L_DIM 2048
#define B_DIM 8
#define N_GROUPS 32
#define CPG 16
#define EPS_GN 1e-6f
#define ATTN_SCALE 0.04419417382415922f  // 1/sqrt(512)

// ---------------- workspace layout (bytes) ----------------
#define OFF_W16 0
#define OFF_HT  (2u * 1024u * 1024u)
#define OFF_QT  (OFF_HT  + 16u * 1024u * 1024u)
#define OFF_KT  (OFF_QT  + 16u * 1024u * 1024u)
#define OFF_V   (OFF_KT  + 16u * 1024u * 1024u)
#define OFF_H2T (OFF_V   + 16u * 1024u * 1024u)
#define OFF_M   (OFF_H2T + 16u * 1024u * 1024u)
#define OFF_Z   (OFF_M   + 64u * 1024u)

#define DEVINL __device__ __forceinline__

// Load a 16x32 (MxK) A fragment -- or equivalently a 32x16 (KxN) B fragment
// whose memory layout is "one matrix row/col per lane, K contiguous".
// CDNA5 16-bit A layout: lanes 0-15 hold K=j (+16 upper chunk), lanes 16-31
// hold K=j+8 (+16): two aligned 16B loads per lane. Works for global or LDS.
DEVINL v16h load_frag(const _Float16* base, int stride) {
  const int lane = threadIdx.x & 31;
  const int idx  = lane & 15;
  const int hi   = lane >> 4;
  const _Float16* p = base + (size_t)idx * (size_t)stride + hi * 8;
  const v8h c0 = *(const v8h*)(p);
  const v8h c1 = *(const v8h*)(p + 16);
  v16h r;
#pragma unroll
  for (int j = 0; j < 8; ++j) { r[j] = c0[j]; r[8 + j] = c1[j]; }
  return r;
}

DEVINL v8f wmma_f16(v16h a, v16h b, v8f c) {
  return __builtin_amdgcn_wmma_f32_16x16x32_f16(
      false, a, false, b, (short)0, c, false, false);
}

// ---------------- kernel 0: fp32 -> fp16 weight conversion ----------------
__global__ void wconv_kernel(const float* __restrict__ Wq, const float* __restrict__ Wk,
                             const float* __restrict__ Wv, const float* __restrict__ Wp,
                             _Float16* __restrict__ W16) {
  int i = blockIdx.x * blockDim.x + threadIdx.x;
  const int n = C_DIM * C_DIM;
  if (i >= 4 * n) return;
  const int sel = i / n, j = i % n;
  const float* src = (sel == 0) ? Wq : (sel == 1) ? Wk : (sel == 2) ? Wv : Wp;
  W16[i] = (_Float16)src[j];
}

// ---------------- kernel 1: GroupNorm, write hT[b][l][c] fp16 ----------------
__global__ void gnorm_kernel(const float* __restrict__ x, const float* __restrict__ gamma,
                             const float* __restrict__ beta, _Float16* __restrict__ hT) {
  const int b = blockIdx.x / N_GROUPS;
  const int g = blockIdx.x % N_GROUPS;
  const int tid = threadIdx.x;
  const int N = CPG * L_DIM;
  const float* xb = x + ((size_t)b * C_DIM + (size_t)g * CPG) * L_DIM;
  float s = 0.f, s2 = 0.f;
  for (int i = tid; i < N; i += 256) {
    float v = xb[i];
    s += v; s2 += v * v;
  }
  __shared__ float r1[256], r2[256];
  r1[tid] = s; r2[tid] = s2;
  __syncthreads();
  for (int off = 128; off > 0; off >>= 1) {
    if (tid < off) { r1[tid] += r1[tid + off]; r2[tid] += r2[tid + off]; }
    __syncthreads();
  }
  const float mean = r1[0] / (float)N;
  const float var  = r2[0] / (float)N - mean * mean;
  const float rstd = rsqrtf(var + EPS_GN);
  for (int i = tid; i < N; i += 256) {
    const int cc = i >> 11;
    const int l  = i & (L_DIM - 1);
    const int c  = g * CPG + cc;
    const float v = (xb[i] - mean) * rstd * gamma[c] + beta[c];
    hT[((size_t)b * L_DIM + l) * C_DIM + c] = (_Float16)v;
  }
}

// ------- kernel 2: QKV projections, 32x32 output per wave (2x2 WMMA tiles) ---
__global__ void qkv_kernel(const _Float16* __restrict__ W16, const _Float16* __restrict__ hT,
                           const float* __restrict__ bq, const float* __restrict__ bk,
                           const float* __restrict__ bv,
                           _Float16* __restrict__ qT, _Float16* __restrict__ kT,
                           _Float16* __restrict__ vCL) {
  int wid = blockIdx.x * (blockDim.x >> 5) + (threadIdx.x >> 5);
  const int lt2 = wid & 63; wid >>= 6;     // 64 l-tiles of 32
  const int ot2 = wid & 15; wid >>= 4;     // 16 o-tiles of 32
  const int b   = wid & 7;  wid >>= 3;
  const int mat = wid;                     // 0=q, 1=k, 2=v
  const int o0 = ot2 * 32, l0 = lt2 * 32;
  const _Float16* A  = W16 + (size_t)mat * C_DIM * C_DIM + (size_t)o0 * C_DIM;
  const _Float16* Bm = hT + ((size_t)b * L_DIM + l0) * C_DIM;
  v8f acc[2][2] = {};
#pragma unroll
  for (int cs = 0; cs < C_DIM / 32; ++cs) {
    const v16h a0 = load_frag(A + cs * 32, C_DIM);
    const v16h a1 = load_frag(A + (size_t)16 * C_DIM + cs * 32, C_DIM);
    const v16h b0 = load_frag(Bm + cs * 32, C_DIM);
    const v16h b1 = load_frag(Bm + (size_t)16 * C_DIM + cs * 32, C_DIM);
    acc[0][0] = wmma_f16(a0, b0, acc[0][0]);
    acc[0][1] = wmma_f16(a0, b1, acc[0][1]);
    acc[1][0] = wmma_f16(a1, b0, acc[1][0]);
    acc[1][1] = wmma_f16(a1, b1, acc[1][1]);
  }
  const int lane = threadIdx.x & 31;
  const int n = lane & 15, hi = lane >> 4;
  const float* bias = (mat == 0) ? bq : (mat == 1) ? bk : bv;
  if (mat < 2) {
    _Float16* dst = (mat == 0) ? qT : kT;
    const float sc = (mat == 0) ? ATTN_SCALE : 1.0f;
#pragma unroll
    for (int i = 0; i < 2; ++i)
#pragma unroll
      for (int j = 0; j < 2; ++j) {
        v8h h;
#pragma unroll
        for (int r = 0; r < 8; ++r) {
          const int o = o0 + i * 16 + hi * 8 + r;   // D row M = r + hi*8
          h[r] = (_Float16)((acc[i][j][r] + bias[o]) * sc);
        }
        *(v8h*)(dst + ((size_t)b * L_DIM + l0 + j * 16 + n) * C_DIM + o0 + i * 16 + hi * 8) = h;
      }
  } else {
#pragma unroll
    for (int i = 0; i < 2; ++i)
#pragma unroll
      for (int j = 0; j < 2; ++j)
#pragma unroll
        for (int r = 0; r < 8; ++r) {
          const int o = o0 + i * 16 + hi * 8 + r;
          vCL[((size_t)b * C_DIM + o) * L_DIM + l0 + j * 16 + n] =
              (_Float16)(acc[i][j][r] + bias[o]);
        }
  }
}

// ---------------- kernel 3: softmax row stats (pass A) ----------------
// One block per (b, 16-query tile). Q tile staged in LDS once (shared by all
// 8 waves); each wave processes pairs of key tiles so one A-frag feeds 2 WMMAs.
__global__ void attn_stats_kernel(const _Float16* __restrict__ qT,
                                  const _Float16* __restrict__ kT,
                                  float* __restrict__ Mbuf, float* __restrict__ Zbuf) {
  const int b  = blockIdx.x >> 7;
  const int lt = blockIdx.x & 127;
  const int l0 = lt * 16;
  const int w    = threadIdx.x >> 5;
  const int lane = threadIdx.x & 31;
  const int hi   = lane >> 4;
  __shared__ __align__(16) _Float16 ldsQ[16 * C_DIM];   // 16 KB
  {
    const v8h* src = (const v8h*)(qT + ((size_t)b * L_DIM + l0) * C_DIM);
    v8h* dst = (v8h*)ldsQ;
    for (int i = threadIdx.x; i < 16 * C_DIM / 8; i += 256) dst[i] = src[i];
  }
  __syncthreads();

  float run_m[8], run_s[8];
#pragma unroll
  for (int r = 0; r < 8; ++r) { run_m[r] = -1e30f; run_s[r] = 0.f; }
  for (int it = 0; it < 8; ++it) {
    const int m0 = (it * 8 + w) * 32;                 // pair of key tiles
    const _Float16* Bk = kT + ((size_t)b * L_DIM + m0) * C_DIM;
    v8f s0 = {}, s1 = {};
#pragma unroll
    for (int cs = 0; cs < C_DIM / 32; ++cs) {
      const v16h a  = load_frag(ldsQ + cs * 32, C_DIM);
      const v16h b0 = load_frag(Bk + cs * 32, C_DIM);
      const v16h b1 = load_frag(Bk + (size_t)16 * C_DIM + cs * 32, C_DIM);
      s0 = wmma_f16(a, b0, s0);
      s1 = wmma_f16(a, b1, s1);
    }
#pragma unroll
    for (int r = 0; r < 8; ++r) {
      float tm = fmaxf(s0[r], s1[r]);
      for (int off = 1; off < 16; off <<= 1) tm = fmaxf(tm, __shfl_xor(tm, off));
      const float nm = fmaxf(run_m[r], tm);
      float pe = __expf(s0[r] - nm) + __expf(s1[r] - nm);
      for (int off = 1; off < 16; off <<= 1) pe += __shfl_xor(pe, off);
      run_s[r] = run_s[r] * __expf(run_m[r] - nm) + pe;
      run_m[r] = nm;
    }
  }
  __shared__ float smax[8][16], ssum[8][16];
  if ((lane & 15) == 0) {
#pragma unroll
    for (int r = 0; r < 8; ++r) { smax[w][hi * 8 + r] = run_m[r]; ssum[w][hi * 8 + r] = run_s[r]; }
  }
  __syncthreads();
  if (threadIdx.x < 16) {
    const int row = threadIdx.x;
    float M = -1e30f;
    for (int ww = 0; ww < 8; ++ww) M = fmaxf(M, smax[ww][row]);
    float Z = 0.f;
    for (int ww = 0; ww < 8; ++ww) Z += ssum[ww][row] * __expf(smax[ww][row] - M);
    Mbuf[(size_t)b * L_DIM + l0 + row] = M;
    Zbuf[(size_t)b * L_DIM + l0 + row] = Z;
  }
}

// ---------------- kernel 4: P = exp(S - M), O = P * V^T (pass B) -------------
// One block per (b, 16-query tile). Q staged in LDS; per iteration the 8 waves
// compute a 16x256 P slab (2 key tiles each) into LDS, then each wave
// accumulates its 64 output channels with 8x4 WMMAs.
__global__ void attn_av_kernel(const _Float16* __restrict__ qT,
                               const _Float16* __restrict__ kT,
                               const _Float16* __restrict__ vCL,
                               const float* __restrict__ Mbuf,
                               const float* __restrict__ Zbuf,
                               _Float16* __restrict__ h2T) {
  const int b  = blockIdx.x >> 7;
  const int lt = blockIdx.x & 127;
  const int l0 = lt * 16;
  const int w    = threadIdx.x >> 5;
  const int lane = threadIdx.x & 31;
  const int n = lane & 15, hi = lane >> 4;
  const int c_base = w * 64;
  __shared__ __align__(16) _Float16 ldsQ[16 * C_DIM];   // 16 KB
  __shared__ __align__(16) _Float16 ldsP[16 * 256];     // 8 KB
  {
    const v8h* src = (const v8h*)(qT + ((size_t)b * L_DIM + l0) * C_DIM);
    v8h* dst = (v8h*)ldsQ;
    for (int i = threadIdx.x; i < 16 * C_DIM / 8; i += 256) dst[i] = src[i];
  }
  float mrow[8], zrow[8];
#pragma unroll
  for (int r = 0; r < 8; ++r) {
    mrow[r] = Mbuf[(size_t)b * L_DIM + l0 + hi * 8 + r];
    zrow[r] = 1.0f / Zbuf[(size_t)b * L_DIM + l0 + hi * 8 + r];
  }
  __syncthreads();

  v8f oacc[4] = {};
  for (int it = 0; it < L_DIM / 256; ++it) {
    const int mb = it * 256;
    const int m0 = mb + w * 32;
    const _Float16* Bk = kT + ((size_t)b * L_DIM + m0) * C_DIM;
    // prefetch next iteration's K rows for this wave (streamed once)
    if (it + 1 < L_DIM / 256)
      __builtin_prefetch(kT + ((size_t)b * L_DIM + m0 + 256) * C_DIM + (size_t)lane * C_DIM, 0, 0);
    v8f s0 = {}, s1 = {};
#pragma unroll
    for (int cs = 0; cs < C_DIM / 32; ++cs) {
      const v16h a  = load_frag(ldsQ + cs * 32, C_DIM);
      const v16h b0 = load_frag(Bk + cs * 32, C_DIM);
      const v16h b1 = load_frag(Bk + (size_t)16 * C_DIM + cs * 32, C_DIM);
      s0 = wmma_f16(a, b0, s0);
      s1 = wmma_f16(a, b1, s1);
    }
#pragma unroll
    for (int r = 0; r < 8; ++r) {
      const int row = hi * 8 + r;                        // D row M = r + hi*8
      ldsP[row * 256 + w * 32 + n]      = (_Float16)__expf(s0[r] - mrow[r]);
      ldsP[row * 256 + w * 32 + 16 + n] = (_Float16)__expf(s1[r] - mrow[r]);
    }
    __syncthreads();
#pragma unroll
    for (int kb = 0; kb < 8; ++kb) {                     // 8 K=32 blocks of 256 keys
      const v16h a = load_frag(ldsP + kb * 32, 256);     // A = P from LDS
#pragma unroll
      for (int ct = 0; ct < 4; ++ct) {                   // wave's 4 channel tiles
        const _Float16* Bv = vCL + ((size_t)b * C_DIM + c_base + (size_t)ct * 16) * L_DIM
                                 + mb + kb * 32;
        oacc[ct] = wmma_f16(a, load_frag(Bv, L_DIM), oacc[ct]);
      }
    }
    __syncthreads();
  }
#pragma unroll
  for (int ct = 0; ct < 4; ++ct)
#pragma unroll
    for (int r = 0; r < 8; ++r)
      h2T[((size_t)b * L_DIM + l0 + hi * 8 + r) * C_DIM + c_base + ct * 16 + n] =
          (_Float16)(oacc[ct][r] * zrow[r]);
}

// -------- kernel 5: out = x + Wp*h2 + bp, 32x32 per wave (2x2 tiles) --------
__global__ void proj_out_kernel(const _Float16* __restrict__ W16,
                                const _Float16* __restrict__ h2T,
                                const float* __restrict__ x,
                                const float* __restrict__ bp,
                                float* __restrict__ out) {
  int wid = blockIdx.x * (blockDim.x >> 5) + (threadIdx.x >> 5);
  const int lt2 = wid & 63; wid >>= 6;
  const int ot2 = wid & 15; wid >>= 4;
  const int b   = wid;
  const int o0 = ot2 * 32, l0 = lt2 * 32;
  const _Float16* A  = W16 + (size_t)3 * C_DIM * C_DIM + (size_t)o0 * C_DIM;   // Wp
  const _Float16* Bm = h2T + ((size_t)b * L_DIM + l0) * C_DIM;
  v8f acc[2][2] = {};
#pragma unroll
  for (int cs = 0; cs < C_DIM / 32; ++cs) {
    const v16h a0 = load_frag(A + cs * 32, C_DIM);
    const v16h a1 = load_frag(A + (size_t)16 * C_DIM + cs * 32, C_DIM);
    const v16h b0 = load_frag(Bm + cs * 32, C_DIM);
    const v16h b1 = load_frag(Bm + (size_t)16 * C_DIM + cs * 32, C_DIM);
    acc[0][0] = wmma_f16(a0, b0, acc[0][0]);
    acc[0][1] = wmma_f16(a0, b1, acc[0][1]);
    acc[1][0] = wmma_f16(a1, b0, acc[1][0]);
    acc[1][1] = wmma_f16(a1, b1, acc[1][1]);
  }
  const int lane = threadIdx.x & 31;
  const int n = lane & 15, hi = lane >> 4;
#pragma unroll
  for (int i = 0; i < 2; ++i)
#pragma unroll
    for (int j = 0; j < 2; ++j)
#pragma unroll
      for (int r = 0; r < 8; ++r) {
        const int o = o0 + i * 16 + hi * 8 + r;
        const size_t idx = ((size_t)b * C_DIM + o) * L_DIM + l0 + j * 16 + n;
        out[idx] = x[idx] + acc[i][j][r] + bp[o];
      }
}

extern "C" void kernel_launch(void* const* d_in, const int* in_sizes, int n_in,
                              void* d_out, int out_size, void* d_ws, size_t ws_size,
                              hipStream_t stream) {
  (void)in_sizes; (void)n_in; (void)out_size; (void)ws_size;
  const float* x     = (const float*)d_in[0];
  const float* Wq    = (const float*)d_in[1];
  const float* bq    = (const float*)d_in[2];
  const float* Wk    = (const float*)d_in[3];
  const float* bk    = (const float*)d_in[4];
  const float* Wv    = (const float*)d_in[5];
  const float* bv    = (const float*)d_in[6];
  const float* Wp    = (const float*)d_in[7];
  const float* bp    = (const float*)d_in[8];
  const float* gamma = (const float*)d_in[9];
  const float* beta  = (const float*)d_in[10];
  float* out = (float*)d_out;

  char* ws = (char*)d_ws;
  _Float16* W16 = (_Float16*)(ws + OFF_W16);
  _Float16* hT  = (_Float16*)(ws + OFF_HT);
  _Float16* qT  = (_Float16*)(ws + OFF_QT);
  _Float16* kT  = (_Float16*)(ws + OFF_KT);
  _Float16* vCL = (_Float16*)(ws + OFF_V);
  _Float16* h2T = (_Float16*)(ws + OFF_H2T);
  float* Mbuf = (float*)(ws + OFF_M);
  float* Zbuf = (float*)(ws + OFF_Z);

  wconv_kernel<<<(4 * C_DIM * C_DIM) / 256, 256, 0, stream>>>(Wq, Wk, Wv, Wp, W16);
  gnorm_kernel<<<B_DIM * N_GROUPS, 256, 0, stream>>>(x, gamma, beta, hT);
  // 3 mats * 8 batches * 16 o-tiles * 64 l-tiles = 24576 waves, 8 waves/block
  qkv_kernel<<<(3 * B_DIM * 16 * 64) / 8, 256, 0, stream>>>(W16, hT, bq, bk, bv, qT, kT, vCL);
  attn_stats_kernel<<<B_DIM * 128, 256, 0, stream>>>(qT, kT, Mbuf, Zbuf);
  attn_av_kernel<<<B_DIM * 128, 256, 0, stream>>>(qT, kT, vCL, Mbuf, Zbuf, h2T);
  // 8 batches * 16 o-tiles * 64 l-tiles = 8192 waves, 8 waves/block
  proj_out_kernel<<<(B_DIM * 16 * 64) / 8, 256, 0, stream>>>(W16, h2T, x, bp, out);
}
